// WindowAttentionT2DSShared_13572096655433
// MI455X (gfx1250) — matrix operand order repaired and checked by reading, not compiled, not run on hardware
//
#include <hip/hip_runtime.h>

#define C_DIM 384
#define C3    1152
#define HEADS 12
#define SCALEF 0.17677669529663687f   // 1/sqrt(32)

typedef __attribute__((ext_vector_type(16))) __bf16 v16bf;
typedef __attribute__((ext_vector_type(8)))  float  v8f;

// ---- helpers -------------------------------------------------------------

__device__ __forceinline__ unsigned short f2b(float f) {
  // float -> bf16 bits, round-to-nearest-even
  unsigned u = __float_as_uint(f);
  u += 0x7FFFu + ((u >> 16) & 1u);
  return (unsigned short)(u >> 16);
}

// A-fragment (16x32 bf16): lane holds row (l&15); lo half-wave K in {0..7,16..23},
// hi half-wave K in {8..15,24..31}.  `base` is row-major bf16 storage.
__device__ __forceinline__ v16bf frag_a(const unsigned short* base, int row,
                                        int stride, int kbase, int hi) {
  const unsigned* q = (const unsigned*)(base + row * stride + kbase + hi * 8);
  union { unsigned u[8]; v16bf v; } t;
  t.u[0] = q[0];  t.u[1] = q[1];  t.u[2] = q[2];  t.u[3] = q[3];
  t.u[4] = q[8];  t.u[5] = q[9];  t.u[6] = q[10]; t.u[7] = q[11];
  return t.v;
}

// B-fragment (32x16 bf16): lane holds column (l&15); lo half-wave K in {0..7,16..23},
// hi half-wave K in {8..15,24..31}.
// `base` is column-major-for-B storage: row index = B column, contiguous K.
__device__ __forceinline__ v16bf frag_b(const unsigned short* base, int row,
                                        int stride, int kbase, int hi) {
  const unsigned* q = (const unsigned*)(base + row * stride + kbase + hi * 8);
  union { unsigned u[8]; v16bf v; } t;
#pragma unroll
  for (int j = 0; j < 8; ++j) t.u[j] = q[(j & 3) + 8 * (j >> 2)];
  return t.v;
}

__device__ __forceinline__ v8f wmma_bf16(v16bf a, v16bf b, v8f c) {
  return __builtin_amdgcn_wmma_f32_16x16x32_bf16(false, a, false, b,
                                                 (short)0, c, false, false);
}

// ---- GEMM: out[M,N] = A[M,K](f32) @ Bw[K,N](f32) + bias, via bf16 WMMA ----
// MODE 0: out -> bf16 bits     MODE 1: out -> f32
// MODE 2: out -> f32, out = resid + alpha[col] * (acc + bias)
template <int MODE>
__global__ __launch_bounds__(256)
void gemm_bf16w(const float* __restrict__ A, const float* __restrict__ Bw,
                const float* __restrict__ bias, int M, int N, int K,
                unsigned short* __restrict__ outB, float* __restrict__ outF,
                const float* __restrict__ resid, const float* __restrict__ alpha) {
  __shared__ unsigned short As[128][32];   // row-major M x K tile
  __shared__ unsigned short Bs[64][32];    // [n][k] (B column-major)

  const int t    = threadIdx.x;
  const int rb   = blockIdx.x * 128;
  const int nb0  = blockIdx.y * 64;
  const int wid  = t >> 5, lane = t & 31;
  const int wm   = wid >> 1, wn = wid & 1;      // 4x2 wave grid of 32x32 tiles
  const int hi   = lane >> 4, ln = lane & 15;

  v8f acc[2][2] = {{{0}}, {{0}}};

  for (int kb = 0; kb < K; kb += 32) {
#pragma unroll
    for (int i = 0; i < 16; ++i) {               // 128x32 A tile
      int e = t + 256 * i;
      int r = e >> 5, c = e & 31;
      As[r][c] = f2b(A[(size_t)(rb + r) * K + kb + c]);
    }
#pragma unroll
    for (int i = 0; i < 8; ++i) {                // 32x64 B tile (transposed in LDS)
      int e = t + 256 * i;
      int kk = e >> 6, n = e & 63;
      Bs[n][kk] = f2b(Bw[(size_t)(kb + kk) * N + nb0 + n]);
    }
    __syncthreads();

    v16bf a0 = frag_a(&As[0][0], wm * 32 + ln,      32, 0, hi);
    v16bf a1 = frag_a(&As[0][0], wm * 32 + 16 + ln, 32, 0, hi);
    v16bf b0 = frag_b(&Bs[0][0], wn * 32 + ln,      32, 0, hi);
    v16bf b1 = frag_b(&Bs[0][0], wn * 32 + 16 + ln, 32, 0, hi);
    acc[0][0] = wmma_bf16(a0, b0, acc[0][0]);
    acc[0][1] = wmma_bf16(a0, b1, acc[0][1]);
    acc[1][0] = wmma_bf16(a1, b0, acc[1][0]);
    acc[1][1] = wmma_bf16(a1, b1, acc[1][1]);
    __syncthreads();
  }

#pragma unroll
  for (int mi = 0; mi < 2; ++mi)
#pragma unroll
    for (int ni = 0; ni < 2; ++ni)
#pragma unroll
      for (int r = 0; r < 8; ++r) {
        int row = rb + wm * 32 + mi * 16 + r + 8 * hi;
        int col = nb0 + wn * 32 + ni * 16 + ln;
        float v = acc[mi][ni][r] + bias[col];
        size_t idx = (size_t)row * N + col;
        if (MODE == 0)      outB[idx] = f2b(v);
        else if (MODE == 1) outF[idx] = v;
        else                outF[idx] = resid[idx] + alpha[col] * v;
      }
}

// ---- Windowed multi-head attention --------------------------------------
// BRANCH 0: (h,w) 7x7 windows, n=49.  BRANCH 1: (d,h) 8x7, n=56.
// BRANCH 2: (d,w) 8x7, n=56.  One block = one (window, head).
// qkv: bf16 bits [M][1152] (q | k | v).  out: f32 [M][384].
template <int BRANCH, bool ACCUM>
__global__ __launch_bounds__(128)
void win_attn(const unsigned short* __restrict__ qkv, float* __restrict__ out,
              int D, int H, int W) {
  const int NV = (BRANCH == 0) ? 49 : 56;

  __shared__ unsigned short Qs[64][32];   // token-major
  __shared__ unsigned short Ks[64][32];   // token-major
  __shared__ unsigned short Vt[32][64];   // dim-major (transposed V)
  __shared__ unsigned short Ps[64][64];   // softmax(P), token-major

  const int t    = threadIdx.x;
  const int head = blockIdx.y;
  const int win  = blockIdx.x;

  // token tk -> global row = rbase + (tk/7)*sA + (tk%7)*sB
  int rbase, sA, sB;
  if (BRANCH == 0) {
    int nWw = W / 7, nWh = H / 7;
    int wb = win % nWw; int r = win / nWw;
    int hb = r % nWh;   r /= nWh;
    int d  = r % D;     int b = r / D;
    rbase = ((b * D + d) * H + hb * 7) * W + wb * 7;
    sA = W; sB = 1;
  } else if (BRANCH == 1) {
    int nWh = H / 7, nDb = D / 8;
    int w  = win % W;  int r = win / W;
    int hb = r % nWh;  r /= nWh;
    int db = r % nDb;  int b = r / nDb;
    rbase = ((b * D + db * 8) * H + hb * 7) * W + w;
    sA = H * W; sB = W;
  } else {
    int nWw = W / 7, nDb = D / 8;
    int wb = win % nWw; int r = win / nWw;
    int h  = r % H;     r /= H;
    int db = r % nDb;   int b = r / nDb;
    rbase = ((b * D + db * 8) * H + h) * W + wb * 7;
    sA = H * W; sB = 1;
  }

  // ---- load Q/K/V for this head (zero-pad tokens >= NV) ----
  const int hoff = head * 32;
#pragma unroll
  for (int m = 0; m < 3; ++m) {
#pragma unroll
    for (int i = 0; i < 8; ++i) {
      int e   = t + 128 * i;          // dword index, 0..1023
      int tok = e >> 4;
      int dp  = e & 15;               // dim pair
      unsigned val = 0u;
      if (tok < NV) {
        int row = rbase + (tok / 7) * sA + (tok % 7) * sB;
        val = *(const unsigned*)(qkv + (size_t)row * C3 + m * C_DIM + hoff + dp * 2);
      }
      if (m == 0)      *(unsigned*)&Qs[tok][dp * 2] = val;
      else if (m == 1) *(unsigned*)&Ks[tok][dp * 2] = val;
      else { Vt[dp * 2][tok]     = (unsigned short)(val & 0xFFFFu);
             Vt[dp * 2 + 1][tok] = (unsigned short)(val >> 16); }
    }
  }
  __syncthreads();

  const int w4 = t >> 5, lane = t & 31;
  const int hi = lane >> 4, ln = lane & 15;

  // ---- S = Q @ K^T  (this wave's 16-row tile vs 4 column tiles) ----
  v16bf aq = frag_a(&Qs[0][0], w4 * 16 + ln, 32, 0, hi);
  v8f s[4];
#pragma unroll
  for (int ct = 0; ct < 4; ++ct) {
    v16bf bk = frag_b(&Ks[0][0], ct * 16 + ln, 32, 0, hi);
    v8f z = {0};
    s[ct] = wmma_bf16(aq, bk, z);
  }

  // ---- row-wise softmax (row lives in a 16-lane half-wave group) ----
#pragma unroll
  for (int r = 0; r < 8; ++r) {
    float mx = -1e30f;
#pragma unroll
    for (int ct = 0; ct < 4; ++ct) {
      float sv = s[ct][r] * SCALEF;
      s[ct][r] = sv;
      if (ct * 16 + ln < NV) mx = fmaxf(mx, sv);
    }
#pragma unroll
    for (int off = 8; off > 0; off >>= 1) mx = fmaxf(mx, __shfl_xor(mx, off, 32));
    float sum = 0.f;
#pragma unroll
    for (int ct = 0; ct < 4; ++ct) {
      float pe = (ct * 16 + ln < NV) ? __expf(s[ct][r] - mx) : 0.f;
      s[ct][r] = pe;
      sum += pe;
    }
#pragma unroll
    for (int off = 8; off > 0; off >>= 1) sum += __shfl_xor(sum, off, 32);
    float inv = 1.f / sum;
#pragma unroll
    for (int ct = 0; ct < 4; ++ct) s[ct][r] *= inv;
  }

  // ---- stash P as bf16 (same-wave LDS RAW kept in order by DScnt) ----
#pragma unroll
  for (int ct = 0; ct < 4; ++ct)
#pragma unroll
    for (int r = 0; r < 8; ++r)
      Ps[w4 * 16 + r + 8 * hi][ct * 16 + ln] = f2b(s[ct][r]);

  // ---- O = P @ V  (16 x 64) @ (64 x 32) ----
#pragma unroll
  for (int nb = 0; nb < 2; ++nb) {
    v8f o = {0};
#pragma unroll
    for (int kc = 0; kc < 2; ++kc) {
      v16bf ap = frag_a(&Ps[0][0], w4 * 16 + ln, 64, kc * 32, hi);
      v16bf bv = frag_b(&Vt[0][0], nb * 16 + ln, 64, kc * 32, hi);
      o = wmma_bf16(ap, bv, o);
    }
#pragma unroll
    for (int r = 0; r < 8; ++r) {
      int tok = w4 * 16 + r + 8 * hi;
      if (tok < NV) {
        int row = rbase + (tok / 7) * sA + (tok % 7) * sB;
        size_t idx = (size_t)row * C_DIM + head * 32 + nb * 16 + ln;
        if (ACCUM) out[idx] += o[r];
        else       out[idx]  = o[r];
      }
    }
  }
}

// ---- host orchestration --------------------------------------------------

extern "C" void kernel_launch(void* const* d_in, const int* in_sizes, int n_in,
                              void* d_out, int out_size, void* d_ws, size_t ws_size,
                              hipStream_t stream) {
  const float* x        = (const float*)d_in[0];
  const float* qkv_w    = (const float*)d_in[1];
  const float* qkv_b    = (const float*)d_in[2];
  const float* qkv_t_w  = (const float*)d_in[3];
  const float* qkv_t_b  = (const float*)d_in[4];
  const float* proj_w   = (const float*)d_in[5];
  const float* proj_b   = (const float*)d_in[6];
  const float* proj_t_w = (const float*)d_in[7];
  const float* proj_t_b = (const float*)d_in[8];
  const float* alpha    = (const float*)d_in[9];

  const int D = 16, H = 56, W = 56;
  const int M = in_sizes[0] / C_DIM;     // B*D*H*W = 100352
  const int B = M / (D * H * W);

  // workspace: bf16 qkv [M][1152] | f32 attn [M][384] | f32 x5 [M][384]
  char* ws = (char*)d_ws;
  unsigned short* qkv = (unsigned short*)ws;
  size_t off = (size_t)M * C3 * sizeof(unsigned short);
  float* attn = (float*)(ws + off);
  off += (size_t)M * C_DIM * sizeof(float);
  float* x5 = (float*)(ws + off);

  dim3 gTall(M / 128, C3 / 64);          // 784 x 18
  dim3 gSq(M / 128, C_DIM / 64);         // 784 x 6

  // 1) qkv = x @ qkv_w + qkv_b  -> bf16
  gemm_bf16w<0><<<gTall, 256, 0, stream>>>(x, qkv_w, qkv_b, M, C3, C_DIM,
                                           qkv, nullptr, nullptr, nullptr);
  // 2) xy-window attention -> attn (f32)
  win_attn<0, false><<<dim3(B * D * (H / 7) * (W / 7), HEADS), 128, 0, stream>>>(
      qkv, attn, D, H, W);
  // 3) x5 = attn @ proj_w + proj_b  -> f32
  gemm_bf16w<1><<<gSq, 256, 0, stream>>>(attn, proj_w, proj_b, M, C_DIM, C_DIM,
                                         nullptr, x5, nullptr, nullptr);
  // 4) qkv_t = x5 @ qkv_t_w + qkv_t_b  -> bf16 (reuse qkv buffer)
  gemm_bf16w<0><<<gTall, 256, 0, stream>>>(x5, qkv_t_w, qkv_t_b, M, C3, C_DIM,
                                           qkv, nullptr, nullptr, nullptr);
  // 5) ty-window attention -> attn (overwrite)
  win_attn<1, false><<<dim3(B * (D / 8) * (H / 7) * W, HEADS), 128, 0, stream>>>(
      qkv, attn, D, H, W);
  // 6) tx-window attention -> attn (+=)
  win_attn<2, true><<<dim3(B * (D / 8) * H * (W / 7), HEADS), 128, 0, stream>>>(
      qkv, attn, D, H, W);
  // 7) out = x5 + alpha * (attn @ proj_t_w + proj_t_b)
  gemm_bf16w<2><<<gSq, 256, 0, stream>>>(attn, proj_t_w, proj_t_b, M, C_DIM, C_DIM,
                                         nullptr, (float*)d_out, x5, alpha);
}
